// FastRCNNModel_10170482557053
// MI455X (gfx1250) — matrix operand order repaired and checked by reading, not compiled, not run on hardware
//
#include <hip/hip_runtime.h>
#include <math.h>

typedef __bf16 bf16;
typedef __attribute__((ext_vector_type(16))) __bf16 v16bf;
typedef __attribute__((ext_vector_type(8)))  float  v8f;

#define NSTRIP 18432   // pixels per im2col strip (multiple of 128)

// ---------------------------------------------------------------------------
// GEMM: C[M][N] = A[M][K] * Bt[N][K]^T  (+ bias, optional ReLU)
// A row-major [M][K], Bt row-major [N][K] (i.e. B column-major).
// K must be a multiple of 32. TA/TB may be float or bf16 (converted in-reg).
// Per-wave tile: 64(M) x 32(N) => 4x2 accumulators of v8f.
// Block: 256 threads = 8 waves arranged 2(M) x 4(N) => 128x128 block tile.
// ---------------------------------------------------------------------------
template <typename TA, typename TB>
__global__ __launch_bounds__(256) void gemm_wmma_kernel(
    const TA* __restrict__ A, const TB* __restrict__ Bt,
    const float* __restrict__ bias, float* __restrict__ C,
    int M, int N, int K, int ldc, int biasPerN, int relu)
{
  const int lane = threadIdx.x & 31;
  const int wave = threadIdx.x >> 5;
  const int wm   = wave & 1;
  const int wn   = wave >> 1;
  const int row0 = blockIdx.y * 128 + wm * 64;
  const int col0 = blockIdx.x * 128 + wn * 32;
  const int lm   = lane & 15;
  const int lh   = lane >> 4;   // 0 or 1

  v8f acc[4][2];
#pragma unroll
  for (int i = 0; i < 4; ++i)
#pragma unroll
    for (int j = 0; j < 2; ++j)
      acc[i][j] = (v8f){0.f, 0.f, 0.f, 0.f, 0.f, 0.f, 0.f, 0.f};

  for (int k0 = 0; k0 < K; k0 += 32) {
    // A fragments: 16x32 bf16. Lane (lm,lh): M=lm, K in {8h..8h+7, 16+8h..16+8h+7}
    v16bf a[4];
#pragma unroll
    for (int i = 0; i < 4; ++i) {
      int m = row0 + 16 * i + lm;
      if (m > M - 1) m = M - 1;              // clamp; stores are guarded
      const TA* p = A + (size_t)m * K + k0 + 8 * lh;
#pragma unroll
      for (int j = 0; j < 8; ++j) a[i][j] = (bf16)(float)p[j];
#pragma unroll
      for (int j = 0; j < 8; ++j) a[i][8 + j] = (bf16)(float)p[16 + j];
    }
    // B fragments: 32x16 bf16. Lane (lm,lh): N=lm, K in {8h..8h+7, 16+8h..16+8h+7}
    v16bf b[2];
#pragma unroll
    for (int j = 0; j < 2; ++j) {
      int n = col0 + 16 * j + lm;
      if (n > N - 1) n = N - 1;
      const TB* p = Bt + (size_t)n * K + k0 + 8 * lh;
#pragma unroll
      for (int t = 0; t < 16; ++t) b[j][t] = (bf16)(float)p[t < 8 ? t : t + 8];
    }
#pragma unroll
    for (int i = 0; i < 4; ++i)
#pragma unroll
      for (int j = 0; j < 2; ++j)
        acc[i][j] = __builtin_amdgcn_wmma_f32_16x16x32_bf16(
            false, a[i], false, b[j], (short)0, acc[i][j], false, false);
  }

  // C/D layout: element (m,n): vgpr r = m%8, lane = (n%16) + 16*(m/8)
#pragma unroll
  for (int i = 0; i < 4; ++i) {
    int mbase = row0 + 16 * i + 8 * lh;
#pragma unroll
    for (int j = 0; j < 2; ++j) {
      int n = col0 + 16 * j + lm;
      if (n < N) {
#pragma unroll
        for (int r = 0; r < 8; ++r) {
          int m = mbase + r;
          if (m < M) {
            float v = acc[i][j][r] + (biasPerN ? bias[n] : bias[m]);
            if (relu && v < 0.f) v = 0.f;
            C[(size_t)m * ldc + n] = v;
          }
        }
      }
    }
  }
}

// ---------------------------------------------------------------------------
// im2col (transposed panel): colBt[p_local][k] = feat[ci][y+kh-1][x+kw-1]
// k = ci*9 + kh*3 + kw ; zero-padded to Kp (multiple of 32) and at borders.
// ---------------------------------------------------------------------------
__global__ void im2col_kernel(const float* __restrict__ feat, bf16* __restrict__ colBt,
                              int Cin, int H, int W, int pStart, int nStrip, int Kp)
{
  size_t idx = (size_t)blockIdx.x * blockDim.x + threadIdx.x;
  size_t total = (size_t)nStrip * Kp;
  if (idx >= total) return;
  int k  = (int)(idx % Kp);
  int pl = (int)(idx / Kp);
  int p  = pStart + pl;
  int y = p / W, x = p % W;
  float v = 0.f;
  int K = Cin * 9;
  if (k < K) {
    int ci = k / 9;
    int r  = k - ci * 9;
    int kh = r / 3, kw = r - kh * 3;
    int yy = y + kh - 1, xx = x + kw - 1;
    if (yy >= 0 && yy < H && xx >= 0 && xx < W)
      v = feat[((size_t)ci * H + yy) * W + xx];
  }
  colBt[idx] = (bf16)v;
}

// Pack conv weights [Cout][Cin][3][3] fp32 -> [Cout][Kp] bf16 (zero-pad K tail)
__global__ void pack_w_kernel(const float* __restrict__ w, bf16* __restrict__ out,
                              int Cout, int K, int Kp)
{
  int idx = blockIdx.x * blockDim.x + threadIdx.x;
  if (idx >= Cout * Kp) return;
  int k = idx % Kp, co = idx / Kp;
  out[idx] = (k < K) ? (bf16)w[(size_t)co * K + k] : (bf16)0.0f;
}

// 2x2 stride-2 max pool, NCHW
__global__ void maxpool_kernel(const float* __restrict__ in, float* __restrict__ out,
                               int C, int H, int W)
{
  int Ho = H >> 1, Wo = W >> 1;
  size_t idx = (size_t)blockIdx.x * blockDim.x + threadIdx.x;
  size_t total = (size_t)C * Ho * Wo;
  if (idx >= total) return;
  int wo = (int)(idx % Wo);
  int ho = (int)((idx / Wo) % Ho);
  int c  = (int)(idx / ((size_t)Wo * Ho));
  const float* p = in + ((size_t)c * H + 2 * ho) * W + 2 * wo;
  out[idx] = fmaxf(fmaxf(p[0], p[1]), fmaxf(p[W], p[W + 1]));
}

// ROI max pool, faithful to reference (rintf = round-half-even, floor/ceil/clip)
__global__ void roi_pool_kernel(const float* __restrict__ feat, const float* __restrict__ regions,
                                float* __restrict__ out, int C, int H, int W)
{
  int idx = blockIdx.x * blockDim.x + threadIdx.x;
  int total = 128 * C * 49;
  if (idx >= total) return;
  int pw = idx % 7;
  int ph = (idx / 7) % 7;
  int c  = (idx / 49) % C;
  int r  = idx / (49 * C);
  const float sc = 1.0f / 16.0f;
  int x1 = (int)rintf(regions[r * 4 + 0] * sc);
  int y1 = (int)rintf(regions[r * 4 + 1] * sc);
  int x2 = (int)rintf(regions[r * 4 + 2] * sc);
  int y2 = (int)rintf(regions[r * 4 + 3] * sc);
  int bwn = x2 - x1 + 1; if (bwn < 1) bwn = 1;
  int bhn = y2 - y1 + 1; if (bhn < 1) bhn = 1;
  float bw = (float)bwn / 7.0f;
  float bh = (float)bhn / 7.0f;
  int ws_ = (int)floorf((float)pw * bw) + x1;
  int we_ = (int)ceilf((float)(pw + 1) * bw) + x1;
  int hs_ = (int)floorf((float)ph * bh) + y1;
  int he_ = (int)ceilf((float)(ph + 1) * bh) + y1;
  ws_ = ws_ < 0 ? 0 : (ws_ > W ? W : ws_);
  we_ = we_ < 0 ? 0 : (we_ > W ? W : we_);
  hs_ = hs_ < 0 ? 0 : (hs_ > H ? H : hs_);
  he_ = he_ < 0 ? 0 : (he_ > H ? H : he_);
  float m = -1e30f;
  for (int y = hs_; y < he_; ++y)
    for (int x = ws_; x < we_; ++x)
      m = fmaxf(m, feat[((size_t)c * H + y) * W + x]);
  if (we_ <= ws_ || he_ <= hs_) m = 0.f;
  out[(size_t)(r * C + c) * 49 + ph * 7 + pw] = m;
}

__global__ void log_softmax_kernel(const float* __restrict__ in, float* __restrict__ out,
                                   int R, int C)
{
  int r = blockIdx.x * blockDim.x + threadIdx.x;
  if (r >= R) return;
  const float* x = in + (size_t)r * C;
  float mx = x[0];
  for (int c = 1; c < C; ++c) mx = fmaxf(mx, x[c]);
  float s = 0.f;
  for (int c = 0; c < C; ++c) s += expf(x[c] - mx);
  float l = logf(s);
  for (int c = 0; c < C; ++c) out[(size_t)r * C + c] = x[c] - mx - l;
}

// ---------------------------------------------------------------------------
extern "C" void kernel_launch(void* const* d_in, const int* in_sizes, int n_in,
                              void* d_out, int out_size, void* d_ws, size_t ws_size,
                              hipStream_t stream)
{
  (void)in_sizes; (void)n_in; (void)out_size; (void)ws_size;

  const float* image   = (const float*)d_in[0];
  const float* regions = (const float*)d_in[1];
  // conv_params: (w_i, b_i) at d_in[2+2i], d_in[3+2i], i = 0..12
  const float* fc1_w = (const float*)d_in[28];
  const float* fc1_b = (const float*)d_in[29];
  const float* fc2_w = (const float*)d_in[30];
  const float* fc2_b = (const float*)d_in[31];
  const float* cls_w = (const float*)d_in[32];
  const float* cls_b = (const float*)d_in[33];
  const float* reg_w = (const float*)d_in[34];
  const float* reg_b = (const float*)d_in[35];

  // ---- workspace carve-up (~165 MB total) ----
  char* wsp = (char*)d_ws;
  auto carve = [&](size_t bytes) -> char* {
    char* p = wsp;
    wsp += (bytes + 255) & ~(size_t)255;
    return p;
  };
  float* actA   = (float*)carve((size_t)64 * 384 * 384 * 4);
  float* actB   = (float*)carve((size_t)64 * 384 * 384 * 4);
  bf16*  colBt  = (bf16*) carve((size_t)42467328);   // max strip panel (1152*18432*2)
  bf16*  packW  = (bf16*) carve((size_t)29421568);   // all packed conv weights
  float* pooled = (float*)carve((size_t)128 * 25088 * 4);
  float* fc1a   = (float*)carve((size_t)128 * 4096 * 4);
  float* fc2a   = (float*)carve((size_t)128 * 4096 * 4);
  float* clsraw = (float*)carve((size_t)128 * 21 * 4);

  static const int cfg[17] = {64, 64, -1, 128, 128, -1, 256, 256, 256, -1,
                              512, 512, 512, -1, 512, 512, 512};

  const float* cur = image;
  float* bufs[2] = {actA, actB};
  int pp = 0;
  int Cin = 3, H = 384, W = 384;
  int li = 0;
  size_t wOff = 0;  // elems into packW

  for (int e = 0; e < 17; ++e) {
    if (cfg[e] < 0) {
      // 2x2 max pool
      float* nxt = bufs[pp];
      int Ho = H >> 1, Wo = W >> 1;
      size_t tot = (size_t)Cin * Ho * Wo;
      maxpool_kernel<<<dim3((unsigned)((tot + 255) / 256)), dim3(256), 0, stream>>>(
          cur, nxt, Cin, H, W);
      cur = nxt; pp ^= 1; H = Ho; W = Wo;
    } else {
      int Cout = cfg[e];
      int K  = Cin * 9;
      int Kp = (K + 31) & ~31;
      const float* wptr = (const float*)d_in[2 + 2 * li];
      const float* bptr = (const float*)d_in[3 + 2 * li];
      bf16* Wp = packW + wOff;
      {
        int tot = Cout * Kp;
        pack_w_kernel<<<dim3((tot + 255) / 256), dim3(256), 0, stream>>>(
            wptr, Wp, Cout, K, Kp);
      }
      int HW = H * W;
      float* nxt = bufs[pp];
      for (int ps = 0; ps < HW; ps += NSTRIP) {
        int ns = HW - ps < NSTRIP ? HW - ps : NSTRIP;
        size_t tot = (size_t)ns * Kp;
        im2col_kernel<<<dim3((unsigned)((tot + 255) / 256)), dim3(256), 0, stream>>>(
            cur, colBt, Cin, H, W, ps, ns, Kp);
        dim3 g((ns + 127) / 128, (Cout + 127) / 128);
        gemm_wmma_kernel<bf16, bf16><<<g, dim3(256), 0, stream>>>(
            Wp, colBt, bptr, nxt + ps, Cout, ns, Kp, HW, /*biasPerN=*/0, /*relu=*/1);
      }
      cur = nxt; pp ^= 1; Cin = Cout; ++li;
      wOff += (size_t)Cout * Kp;
    }
  }

  // cur = final feature map [512][24][24]
  {
    int tot = 128 * 512 * 49;
    roi_pool_kernel<<<dim3((tot + 255) / 256), dim3(256), 0, stream>>>(
        cur, regions, pooled, 512, 24, 24);
  }

  // FC layers: weights already in [N][K] layout, read fp32 and convert in-register
  gemm_wmma_kernel<float, float><<<dim3(4096 / 128, 1), dim3(256), 0, stream>>>(
      pooled, fc1_w, fc1_b, fc1a, 128, 4096, 25088, 4096, /*biasPerN=*/1, /*relu=*/1);
  gemm_wmma_kernel<float, float><<<dim3(4096 / 128, 1), dim3(256), 0, stream>>>(
      fc1a, fc2_w, fc2_b, fc2a, 128, 4096, 4096, 4096, 1, 1);
  gemm_wmma_kernel<float, float><<<dim3(1, 1), dim3(256), 0, stream>>>(
      fc2a, cls_w, cls_b, clsraw, 128, 21, 4096, 21, 1, 0);

  float* out_cls = (float*)d_out;              // 128*21
  float* out_reg = (float*)d_out + 128 * 21;   // 128*84
  gemm_wmma_kernel<float, float><<<dim3(1, 1), dim3(256), 0, stream>>>(
      fc2a, reg_w, reg_b, out_reg, 128, 84, 4096, 84, 1, 0);

  log_softmax_kernel<<<dim3(1), dim3(128), 0, stream>>>(clsraw, out_cls, 128, 21);
}
